// GCN_85091892068582
// MI455X (gfx1250) — matrix operand order repaired and checked by reading, not compiled, not run on hardware
//
#include <hip/hip_runtime.h>
#include <hip/hip_bf16.h>
#include <math.h>

// ---------------------------------------------------------------------------
// GCN forward on MI455X (gfx1250, wave32).
//   deg -> norm(rsqrt)  -> [scale, scatter-add, scale, GEMM(bf16 WMMA), BN+ReLU] x2
//   -> GEMM3 (project first, out-scaled by norm) -> scatter-add -> +b3 -> log_softmax
// Workspace layout (f32): A[M*256] | B[M*256] | norm[M] | sums[256] | sumsq[256]
//                         | scale[256] | shift[256]   (~205 MB for M=100000)
// ---------------------------------------------------------------------------

typedef __attribute__((ext_vector_type(16))) __bf16 v16bf;
typedef __attribute__((ext_vector_type(8)))  float  v8f;

#define DEV_ATOMIC_ADD(p, v) \
  __hip_atomic_fetch_add((p), (v), __ATOMIC_RELAXED, __HIP_MEMORY_SCOPE_AGENT)

// ------------------------------- utility -----------------------------------
__global__ void k_zero(float* __restrict__ p, long long n) {
  long long i = (long long)blockIdx.x * blockDim.x + threadIdx.x;
  if (i < n) p[i] = 0.0f;
}

__global__ void k_degree(const int* __restrict__ dst, float* __restrict__ deg, int E) {
  int e = blockIdx.x * blockDim.x + threadIdx.x;
  if (e < E) {
    __builtin_prefetch(dst + e + 2048, 0, 0);   // global_prefetch_b8
    DEV_ATOMIC_ADD(&deg[dst[e]], 1.0f);
  }
}

__global__ void k_norm(float* __restrict__ nrm, int M) {
  int i = blockIdx.x * blockDim.x + threadIdx.x;
  if (i < M) nrm[i] = rsqrtf(fmaxf(nrm[i], 1.0f));
}

// out[r][c] = x[r][c] * nrm[r]
template <int C>
__global__ void k_rowscale(const float* __restrict__ x, const float* __restrict__ nrm,
                           float* __restrict__ o, int M) {
  long long i = (long long)blockIdx.x * blockDim.x + threadIdx.x;
  if (i < (long long)M * C) {
    int r = (int)(i / C);
    o[i] = x[i] * nrm[r];
  }
}

// ------------------------- edge scatter-add (f32 atomics) -------------------
// One float4 per thread; CH = F/4 chunks per edge; a wave covers consecutive
// chunks of one edge -> coalesced gather + contiguous atomic adds (L2-resident).
template <int F, int CH>
__global__ void k_scatter_add(const float* __restrict__ H, const int* __restrict__ src,
                              const int* __restrict__ dst, float* __restrict__ Mo, int E) {
  long long tid = (long long)blockIdx.x * blockDim.x + threadIdx.x;
  if (tid >= (long long)E * CH) return;
  int e = (int)(tid / CH);
  int c = (int)(tid % CH) * 4;
  __builtin_prefetch(src + e + 4096, 0, 0);
  int s = src[e], d = dst[e];
  const float4 v = *(const float4*)(H + (size_t)s * F + c);
  float* p = Mo + (size_t)d * F + c;
  DEV_ATOMIC_ADD(p + 0, v.x);
  DEV_ATOMIC_ADD(p + 1, v.y);
  DEV_ATOMIC_ADD(p + 2, v.z);
  DEV_ATOMIC_ADD(p + 3, v.w);
}

// ------------------------------ WMMA GEMM ----------------------------------
// D[M x NW] = (SIN ? diag(nrm) : I) * Af[M x K] @ Wf[K x NW]  (+bias) (* nrm if SOUT)
// 2 waves per block: two adjacent 16-row M-tiles sharing one 16-col N-tile
// (W fragments reused through L0/L1). K stepped by 32 with
// v_wmma_f32_16x16x32_bf16. Fragment layouts per CDNA5 ISA 7.12.2 (wave32):
//   A 16x32 bf16: lane<16 -> elems 0..7 = K0..7, 8..15 = K16..23 ; lane>=16 -> +8
//   B 32x16 bf16: N = lane&15, same K order as A (elems 0..7 = K0..7, 8..15 = K16..23; lane>=16 -> +8)
//   C/D f32:      N = lane&15, vgpr j -> Mrow = j + 8*(lane>>4)
// Padding trick: B column n only affects D column n, and each lane's acc holds
// only its own column, so out-of-range columns (layer 3: 40..47) just CLAMP the
// load address (unconditional coalesced loads, no exec-mask divergence); their
// garbage results are simply never stored.
template <int K, int NW, bool SIN, bool SOUT, bool BIAS>
__global__ void k_wmma_gemm(const float* __restrict__ Af, const float* __restrict__ Wf,
                            const float* __restrict__ bias, const float* __restrict__ nrm,
                            float* __restrict__ D, int M) {
  const int lane = threadIdx.x & 31;
  const int wid  = threadIdx.x >> 5;      // 0..1
  const int half = lane >> 4;
  const int l15  = lane & 15;
  const int row0 = (blockIdx.x * 2 + wid) * 16;
  if (row0 >= M) return;
  const int col0 = blockIdx.y * 16;

  const int arow = row0 + l15;
  const int bcol = col0 + l15;
  const int bcolc = (bcol < NW) ? bcol : 0;       // clamped, branchless
  const float rs = SIN ? nrm[arow] : 1.0f;

  const float* ap0 = Af + (size_t)arow * K + 8 * half;     // + k0 each iter
  const float* wp0 = Wf + (size_t)(8 * half) * NW + bcolc; // + k0*NW each iter

  v8f acc = {};
  for (int k0 = 0; k0 < K; k0 += 32) {
    // ---- A fragment (row arow, on-the-fly bf16 + fused norm row-scale) ----
    const float* ap = ap0 + k0;
    v16bf a;
#pragma unroll
    for (int i = 0; i < 8; ++i) a[i] = (__bf16)(ap[i] * rs);
#pragma unroll
    for (int i = 0; i < 8; ++i) a[8 + i] = (__bf16)(ap[16 + i] * rs);
    // ---- B fragment: 16 unconditional coalesced b32 loads, stride NW ----
    const float* wp = wp0 + (size_t)k0 * NW;
    v16bf b;
#pragma unroll
    for (int i = 0; i < 16; ++i) b[i] = (__bf16)wp[(size_t)(i + (i >= 8 ? 8 : 0)) * NW];
    acc = __builtin_amdgcn_wmma_f32_16x16x32_bf16(false, a, false, b,
                                                  (short)0, acc, false, false);
  }

  if (bcol < NW) {
    const float bv = BIAS ? bias[bcol] : 0.0f;
#pragma unroll
    for (int j = 0; j < 8; ++j) {
      int row = row0 + j + 8 * half;
      float os = SOUT ? nrm[row] : 1.0f;
      D[(size_t)row * NW + bcol] = acc[j] * os + bv;
    }
  }
}

// ------------------------------ BatchNorm ----------------------------------
// Per-column partial sums: thread = column, block iterates a row range.
__global__ void k_bn_stats(const float* __restrict__ H, int M,
                           float* __restrict__ sum, float* __restrict__ sumsq) {
  const int C = 256;
  int c = threadIdx.x;
  int rows_per = (M + gridDim.x - 1) / gridDim.x;
  int r0 = blockIdx.x * rows_per;
  int r1 = min(M, r0 + rows_per);
  float s = 0.0f, q = 0.0f;
  for (int r = r0; r < r1; ++r) {
    float v = H[(size_t)r * C + c];
    s += v; q += v * v;
  }
  DEV_ATOMIC_ADD(&sum[c], s);
  DEV_ATOMIC_ADD(&sumsq[c], q);
}

__global__ void k_bn_finalize(const float* __restrict__ sum, const float* __restrict__ sumsq,
                              const float* __restrict__ gamma, const float* __restrict__ beta,
                              float* __restrict__ scl, float* __restrict__ shf, int M) {
  int c = threadIdx.x;  // 256 threads, one block
  float inv_m = 1.0f / (float)M;
  float mean = sum[c] * inv_m;
  float var  = sumsq[c] * inv_m - mean * mean;
  float istd = rsqrtf(var + 1e-5f);
  float s = istd * gamma[c];
  scl[c] = s;
  shf[c] = beta[c] - mean * s;
}

// out = relu(H*scale + shift) [* nrm if SC]
template <bool SC>
__global__ void k_bn_apply(const float* __restrict__ H, const float* __restrict__ scl,
                           const float* __restrict__ shf, const float* __restrict__ nrm,
                           float* __restrict__ O, int M) {
  const int C = 256;
  long long i = (long long)blockIdx.x * blockDim.x + threadIdx.x;
  if (i < (long long)M * C) {
    int r = (int)(i / C);
    int c = (int)(i % C);
    float v = fmaxf(H[i] * scl[c] + shf[c], 0.0f);
    O[i] = SC ? v * nrm[r] : v;
  }
}

// --------------------------- final log-softmax ------------------------------
// One wave32 per row of 40 logits: logits = M3[r]*nrm[r] + b3, then log_softmax.
__global__ void k_logsoftmax(const float* __restrict__ M3, const float* __restrict__ nrm,
                             const float* __restrict__ b3, float* __restrict__ out, int M) {
  const int NC = 40;
  int wave = threadIdx.x >> 5;
  int lane = threadIdx.x & 31;
  int row  = blockIdx.x * (blockDim.x >> 5) + wave;
  if (row >= M) return;
  float nr = nrm[row];
  const float* mp = M3 + (size_t)row * NC;
  int c0 = lane, c1 = lane + 32;
  float v0 = mp[c0] * nr + b3[c0];                       // c0 < 40 always (lane<=31)
  float v1 = (c1 < NC) ? (mp[c1] * nr + b3[c1]) : -INFINITY;
  float mx = fmaxf(v0, v1);
#pragma unroll
  for (int o = 16; o > 0; o >>= 1) mx = fmaxf(mx, __shfl_xor(mx, o));
  float s = __expf(v0 - mx) + ((c1 < NC) ? __expf(v1 - mx) : 0.0f);
#pragma unroll
  for (int o = 16; o > 0; o >>= 1) s += __shfl_xor(s, o);
  float lse = mx + __logf(s);
  float* op = out + (size_t)row * NC;
  op[c0] = v0 - lse;
  if (c1 < NC) op[c1] = v1 - lse;
}

// ------------------------------- launcher -----------------------------------
static inline int blocks_for(long long n, int bs) { return (int)((n + bs - 1) / bs); }

extern "C" void kernel_launch(void* const* d_in, const int* in_sizes, int n_in,
                              void* d_out, int out_size, void* d_ws, size_t ws_size,
                              hipStream_t stream) {
  const float* x      = (const float*)d_in[0];
  const int*   src    = (const int*)  d_in[1];
  const int*   dst    = (const int*)  d_in[2];
  const float* W1     = (const float*)d_in[3];
  const float* b1     = (const float*)d_in[4];
  const float* gamma1 = (const float*)d_in[5];
  const float* beta1  = (const float*)d_in[6];
  const float* W2     = (const float*)d_in[7];
  const float* b2     = (const float*)d_in[8];
  const float* gamma2 = (const float*)d_in[9];
  const float* beta2  = (const float*)d_in[10];
  const float* W3     = (const float*)d_in[11];
  const float* b3     = (const float*)d_in[12];
  float*       out    = (float*)d_out;

  const int M = in_sizes[0] / 128;   // 100000 nodes
  const int E = in_sizes[1];         // 3.2M directed edges

  float* A    = (float*)d_ws;        // M*256
  float* B    = A + (size_t)M * 256; // M*256
  float* nrm  = B + (size_t)M * 256; // M
  float* sums = nrm + M;             // 256
  float* sumq = sums + 256;          // 256
  float* scl  = sumq + 256;          // 256
  float* shf  = scl + 256;           // 256

  const int BS = 256;
  const int mtiles2 = (M / 16 + 1) / 2;   // 2 M-tiles per block

  // ---- degree -> norm ----
  k_zero<<<blocks_for(M, BS), BS, 0, stream>>>(nrm, M);
  k_degree<<<blocks_for(E, BS), BS, 0, stream>>>(dst, nrm, E);
  k_norm<<<blocks_for(M, BS), BS, 0, stream>>>(nrm, M);

  // ---- layer 1: aggregate(128) then GEMM 128->256 ----
  k_rowscale<128><<<blocks_for((long long)M * 128, BS), BS, 0, stream>>>(x, nrm, A, M);
  k_zero<<<blocks_for((long long)M * 128, BS), BS, 0, stream>>>(B, (long long)M * 128);
  k_scatter_add<128, 32><<<blocks_for((long long)E * 32, BS), BS, 0, stream>>>(A, src, dst, B, E);
  k_wmma_gemm<128, 256, true, false, true>
      <<<dim3(mtiles2, 16), 64, 0, stream>>>(B, W1, b1, nrm, A, M);

  // BN1 + ReLU, fused with next layer's input row-scale
  k_zero<<<blocks_for(512, BS), BS, 0, stream>>>(sums, 512);
  k_bn_stats<<<512, 256, 0, stream>>>(A, M, sums, sumq);
  k_bn_finalize<<<1, 256, 0, stream>>>(sums, sumq, gamma1, beta1, scl, shf, M);
  k_bn_apply<true><<<blocks_for((long long)M * 256, BS), BS, 0, stream>>>(A, scl, shf, nrm, B, M);

  // ---- layer 2: aggregate(256) then GEMM 256->256 ----
  k_zero<<<blocks_for((long long)M * 256, BS), BS, 0, stream>>>(A, (long long)M * 256);
  k_scatter_add<256, 64><<<blocks_for((long long)E * 64, BS), BS, 0, stream>>>(B, src, dst, A, E);
  k_wmma_gemm<256, 256, true, false, true>
      <<<dim3(mtiles2, 16), 64, 0, stream>>>(A, W2, b2, nrm, B, M);

  // BN2 + ReLU (no norm scale: layer 3 projects first)
  k_zero<<<blocks_for(512, BS), BS, 0, stream>>>(sums, 512);
  k_bn_stats<<<512, 256, 0, stream>>>(B, M, sums, sumq);
  k_bn_finalize<<<1, 256, 0, stream>>>(sums, sumq, gamma2, beta2, scl, shf, M);
  k_bn_apply<false><<<blocks_for((long long)M * 256, BS), BS, 0, stream>>>(B, scl, shf, nrm, A, M);

  // ---- layer 3: GEMM 256->40 (padded to 48, out-scaled by norm), aggregate(40) ----
  k_wmma_gemm<256, 40, false, true, false>
      <<<dim3(mtiles2, 3), 64, 0, stream>>>(A, W3, nullptr, nrm, B, M);
  k_zero<<<blocks_for((long long)M * 40, BS), BS, 0, stream>>>(A, (long long)M * 40);
  k_scatter_add<40, 10><<<blocks_for((long long)E * 10, BS), BS, 0, stream>>>(B, src, dst, A, E);

  // ---- +b3, log_softmax ----
  k_logsoftmax<<<blocks_for(M, 8), 256, 0, stream>>>(A, nrm, b3, out, M);
}
